// MoEBlock_20581483283090
// MI455X (gfx1250) — matrix operand order repaired and checked by reading, not compiled, not run on hardware
//
#include <hip/hip_runtime.h>
#include <hip/hip_bf16.h>
#include <math.h>

// ---------------- problem constants ----------------
#define Bb   4
#define Nn   1024
#define Cc   512
#define Hh   8
#define Ee   8
#define TOPK 2
#define HDm  64      // Cc/Hh
#define MLPF 2048    // 4*Cc
#define NTOK (Bb*Nn) // 4096
#define SCALE_ 8.0f  // sqrt(head_dim), bug kept from reference
#define REG_   0.01f
#define EPS_   1e-5f

typedef unsigned short u16;
typedef __bf16 bf16_t;
typedef bf16_t v16bf __attribute__((ext_vector_type(16)));
typedef float  v8f   __attribute__((ext_vector_type(8)));

struct alignas(16) B16v { unsigned int a, b, c, d; };
union FragU { v16bf v; B16v q[2]; };

__device__ inline u16 f2bf(float f) {
  unsigned int u = __float_as_uint(f);
  u += 0x7fffu + ((u >> 16) & 1u);   // round-to-nearest-even
  return (u16)(u >> 16);
}

// ---------------- init: zero combine output + router stats ----------------
__global__ void init_kernel(float* __restrict__ out, float* __restrict__ stats) {
  long i = (long)blockIdx.x * 256 + threadIdx.x;
  if (i < (long)NTOK * Cc) out[i] = 0.0f;
  if (i < 2 * Ee) stats[i] = 0.0f;
}

// ---------------- fp32 -> bf16 elementwise ----------------
__global__ void f2bf_kernel(const float* __restrict__ s, u16* __restrict__ d, long n) {
  long i = (long)blockIdx.x * 256 + threadIdx.x;
  if (i < n) d[i] = f2bf(s[i]);
}

// ---------------- router: logits, softmax, top-2 gates, stats ----------------
__global__ void router_kernel(const float* __restrict__ x, const float* __restrict__ rW,
                              const float* __restrict__ rb, const float* __restrict__ noise,
                              float* __restrict__ gate, float* __restrict__ counts,
                              float* __restrict__ probsum) {
  int t = blockIdx.x * blockDim.x + threadIdx.x;
  if (t >= NTOK) return;
  const float* xr = x + (long)t * Cc;
  float logit[Ee];
#pragma unroll
  for (int e = 0; e < Ee; e++) {
    const float* w = rW + (long)e * Cc;
    float s = 0.0f;
    for (int c = 0; c < Cc; c++) s += xr[c] * w[c];
    logit[e] = s + rb[e] + noise[(long)t * Ee + e] * 0.1f;
  }
  float mx = logit[0];
#pragma unroll
  for (int e = 1; e < Ee; e++) mx = fmaxf(mx, logit[e]);
  float p[Ee], ps = 0.0f;
#pragma unroll
  for (int e = 0; e < Ee; e++) { p[e] = __expf(logit[e] - mx); ps += p[e]; }
#pragma unroll
  for (int e = 0; e < Ee; e++) p[e] /= ps;
  int i1 = 0;
#pragma unroll
  for (int e = 1; e < Ee; e++) if (p[e] > p[i1]) i1 = e;
  int i2 = (i1 == 0) ? 1 : 0;
#pragma unroll
  for (int e = 0; e < Ee; e++) if (e != i1 && p[e] > p[i2]) i2 = e;
  float s2 = p[i1] + p[i2];
#pragma unroll
  for (int e = 0; e < Ee; e++) gate[(long)t * Ee + e] = 0.0f;
  gate[(long)t * Ee + i1] = p[i1] / s2;
  gate[(long)t * Ee + i2] = p[i2] / s2;
  atomicAdd(&counts[i1], 1.0f);
  atomicAdd(&counts[i2], 1.0f);
#pragma unroll
  for (int e = 0; e < Ee; e++) atomicAdd(&probsum[e], p[e]);
}

// ---------------- LayerNorm row (C=512) -> bf16 ----------------
__global__ void ln_kernel(const float* __restrict__ x, const float* __restrict__ g,
                          const float* __restrict__ b, u16* __restrict__ out) {
  int row = blockIdx.x, tid = threadIdx.x;   // 256 threads
  const float* xr = x + (long)row * Cc;
  float v0 = xr[tid], v1 = xr[tid + 256];
  __shared__ float s1[256], s2[256];
  s1[tid] = v0 + v1;
  s2[tid] = v0 * v0 + v1 * v1;
  __syncthreads();
  for (int s = 128; s > 0; s >>= 1) {
    if (tid < s) { s1[tid] += s1[tid + s]; s2[tid] += s2[tid + s]; }
    __syncthreads();
  }
  float mean = s1[0] * (1.0f / Cc);
  float var  = s2[0] * (1.0f / Cc) - mean * mean;
  float inv  = rsqrtf(var + EPS_);
  out[(long)row * Cc + tid]       = f2bf((v0 - mean) * inv * g[tid] + b[tid]);
  out[(long)row * Cc + tid + 256] = f2bf((v1 - mean) * inv * g[tid + 256] + b[tid + 256]);
}

// ---------------- split qkv[f32 B,N,3,H,HD] -> q,k [B,H,N,HD] bf16 + v^T [B,H,HD,N] bf16 ----------------
__global__ void split_qkv_kernel(const float* __restrict__ qkv, u16* __restrict__ qb,
                                 u16* __restrict__ kb, u16* __restrict__ vt) {
  long i = (long)blockIdx.x * 256 + threadIdx.x;
  if (i >= (long)NTOK * Cc) return;
  int d = (int)(i % HDm);
  int h = (int)((i / HDm) % Hh);
  int n = (int)((i / Cc) % Nn);
  int b = (int)(i / ((long)Nn * Cc));
  const float* base = qkv + ((long)(b * Nn + n) * 3) * Cc + h * HDm + d;
  long bh = (long)(b * Hh + h);
  qb[(bh * Nn + n) * HDm + d] = f2bf(base[0]);
  kb[(bh * Nn + n) * HDm + d] = f2bf(base[Cc]);
  vt[(bh * HDm + d) * Nn + n] = f2bf(base[2 * Cc]);
}

// ---------------- row softmax over N=1024, write bf16 in-place (first half of f32 row) ----------------
__global__ void softmax_kernel(float* __restrict__ S) {
  long row = blockIdx.x;
  float* r = S + row * (long)Nn;
  u16*   o = ((u16*)S) + row * (long)(2 * Nn);
  int tid = threadIdx.x;
  float v[4];
#pragma unroll
  for (int i = 0; i < 4; i++) v[i] = r[tid + i * 256];
  __shared__ float sm[256];
  float mx = fmaxf(fmaxf(v[0], v[1]), fmaxf(v[2], v[3]));
  sm[tid] = mx; __syncthreads();
  for (int s = 128; s > 0; s >>= 1) { if (tid < s) sm[tid] = fmaxf(sm[tid], sm[tid + s]); __syncthreads(); }
  mx = sm[0]; __syncthreads();
  float e[4], ls = 0.0f;
#pragma unroll
  for (int i = 0; i < 4; i++) { e[i] = __expf(v[i] - mx); ls += e[i]; }
  sm[tid] = ls; __syncthreads();
  for (int s = 128; s > 0; s >>= 1) { if (tid < s) sm[tid] += sm[tid + s]; __syncthreads(); }
  float inv = 1.0f / sm[0];
#pragma unroll
  for (int i = 0; i < 4; i++) o[tid + i * 256] = f2bf(e[i] * inv);
}

// ---------------- generic batched WMMA GEMM ----------------
// C[z][M,Ncols] = act( alpha * A[z][M,K](bf16) * B[z][Ncols,K]^T(bf16) + bias[N] + resid )
// z base offsets: (z/zdiv)*s1 + (z%zdiv)*s2  (element units)
// REQUIREMENT: M % 16 == 0, Ncols % 64 == 0, Kdim % 32 == 0 (true for all launches here)
__global__ void __launch_bounds__(256)
wmma_gemm_kernel(const u16* __restrict__ A, long lda, long saz1, long saz2,
                 const u16* __restrict__ Bm, long ldb, long sbz1, long sbz2,
                 const float* __restrict__ bias,
                 const float* __restrict__ resid, long ldr, long srz1, long srz2,
                 float* __restrict__ Cf, u16* __restrict__ Cbf, long ldc, long scz1, long scz2,
                 int M, int Ncols, int Kdim, int zdiv, float alpha, int act) {
  int z = blockIdx.z;
  long abase = (long)(z / zdiv) * saz1 + (long)(z % zdiv) * saz2;
  long bbase = (long)(z / zdiv) * sbz1 + (long)(z % zdiv) * sbz2;
  long cbase = (long)(z / zdiv) * scz1 + (long)(z % zdiv) * scz2;
  long rbase = (long)(z / zdiv) * srz1 + (long)(z % zdiv) * srz2;

  int wave = threadIdx.x >> 5;
  int lane = threadIdx.x & 31;
  int hi   = lane >> 4;       // K-half select
  int r    = lane & 15;       // row (A) / column (B,D)

  int mrow  = blockIdx.y * 128 + wave * 16;
  if (mrow >= M) return;                       // uniform per wave
  int ncol0 = blockIdx.x * 64;

  v8f acc0 = {}, acc1 = {}, acc2 = {}, acc3 = {};
  const u16* Arow = A + abase + (long)(mrow + r) * lda;
  // B row base for this lane's column within each of the 4 subtiles
  const u16* Brow0 = Bm + bbase + (long)(ncol0 + 0 * 16 + r) * ldb + hi * 16;
  const u16* Brow1 = Bm + bbase + (long)(ncol0 + 1 * 16 + r) * ldb + hi * 16;
  const u16* Brow2 = Bm + bbase + (long)(ncol0 + 2 * 16 + r) * ldb + hi * 16;
  const u16* Brow3 = Bm + bbase + (long)(ncol0 + 3 * 16 + r) * ldb + hi * 16;

  for (int k = 0; k < Kdim; k += 32) {
    // A 16x32 bf16 frag: lane r (hi=0): K k..k+15; hi=1: K k+16..k+31 (contiguous, matches B)
    FragU fa, fb0, fb1, fb2, fb3;
    int ka = k + (hi ? 16 : 0);
    fa.q[0]  = *(const B16v*)(Arow + ka);
    fa.q[1]  = *(const B16v*)(Arow + ka + 8);
    // B 32x16 bf16 frags: lane r = column, hi=0 -> K k..k+15, hi=1 -> K k+16..k+31 (contiguous 32B)
    fb0.q[0] = *(const B16v*)(Brow0 + k);
    fb0.q[1] = *(const B16v*)(Brow0 + k + 8);
    fb1.q[0] = *(const B16v*)(Brow1 + k);
    fb1.q[1] = *(const B16v*)(Brow1 + k + 8);
    fb2.q[0] = *(const B16v*)(Brow2 + k);
    fb2.q[1] = *(const B16v*)(Brow2 + k + 8);
    fb3.q[0] = *(const B16v*)(Brow3 + k);
    fb3.q[1] = *(const B16v*)(Brow3 + k + 8);
    acc0 = __builtin_amdgcn_wmma_f32_16x16x32_bf16(false, fa.v, false, fb0.v, (short)0, acc0, false, false);
    acc1 = __builtin_amdgcn_wmma_f32_16x16x32_bf16(false, fa.v, false, fb1.v, (short)0, acc1, false, false);
    acc2 = __builtin_amdgcn_wmma_f32_16x16x32_bf16(false, fa.v, false, fb2.v, (short)0, acc2, false, false);
    acc3 = __builtin_amdgcn_wmma_f32_16x16x32_bf16(false, fa.v, false, fb3.v, (short)0, acc3, false, false);
  }

  // D layout: lane r column n, hi selects M rows 0..7 / 8..15 across the 8 acc VGPRs
  int mb = mrow + hi * 8;
  v8f accs[4] = { acc0, acc1, acc2, acc3 };
#pragma unroll
  for (int j = 0; j < 4; j++) {
    int n = ncol0 + j * 16 + r;
    float bs = bias ? bias[n] : 0.0f;
#pragma unroll
    for (int v = 0; v < 8; v++) {
      float val = accs[j][v] * alpha + bs;
      long ro = (long)(mb + v);
      if (resid) val += resid[rbase + ro * ldr + n];
      if (act == 1) val = 0.5f * val * (1.0f + erff(val * 0.70710678118654752f));
      long co = cbase + ro * ldc + n;
      if (Cbf) Cbf[co] = f2bf(val);
      else     Cf[co]  = val;
    }
  }
}

// ---------------- gated accumulate into combine output ----------------
__global__ void combine_kernel(const float* __restrict__ eo, const float* __restrict__ gate,
                               float* __restrict__ out, int e) {
  long i = (long)blockIdx.x * 256 + threadIdx.x;
  if (i >= (long)NTOK * Cc) return;
  long t = i / Cc;
  out[i] += gate[t * Ee + e] * eo[i];
}

// ---------------- balance loss scalar ----------------
__global__ void balance_kernel(const float* __restrict__ rW, const float* __restrict__ stats,
                               float* __restrict__ out) {
  __shared__ float sh[256];
  int tid = threadIdx.x;
  float s = 0.0f;
  for (int i = tid; i < Ee * Cc; i += 256) s += rW[i] * rW[i];
  sh[tid] = s; __syncthreads();
  for (int k = 128; k > 0; k >>= 1) { if (tid < k) sh[tid] += sh[tid + k]; __syncthreads(); }
  if (tid == 0) {
    float bal = 0.0f;
    for (int e = 0; e < Ee; e++)
      bal += (stats[e] / (float)(NTOK * TOPK)) * (stats[Ee + e] / (float)NTOK);
    out[(long)NTOK * Cc] = (float)Ee * bal + REG_ * sqrtf(sh[0]);
  }
}

// ---------------- host helper ----------------
static void launch_gemm(hipStream_t stream,
                        const u16* A, long lda, long saz1, long saz2,
                        const u16* Bm, long ldb, long sbz1, long sbz2,
                        const float* bias,
                        const float* resid, long ldr, long srz1, long srz2,
                        float* Cf, u16* Cbf, long ldc, long scz1, long scz2,
                        int M, int Ncols, int Kdim, int gz, int zdiv, float alpha, int act) {
  dim3 grid((Ncols + 63) / 64, (M + 127) / 128, gz);
  wmma_gemm_kernel<<<grid, 256, 0, stream>>>(A, lda, saz1, saz2, Bm, ldb, sbz1, sbz2, bias,
                                             resid, ldr, srz1, srz2, Cf, Cbf, ldc, scz1, scz2,
                                             M, Ncols, Kdim, zdiv, alpha, act);
}

extern "C" void kernel_launch(void* const* d_in, const int* in_sizes, int n_in,
                              void* d_out, int out_size, void* d_ws, size_t ws_size,
                              hipStream_t stream) {
  (void)in_sizes; (void)n_in; (void)out_size; (void)ws_size;
  const float* x     = (const float*)d_in[0];
  const float* noise = (const float*)d_in[1];
  const float* rW    = (const float*)d_in[2];
  const float* rb    = (const float*)d_in[3];
  const float* ln1g  = (const float*)d_in[4];
  const float* ln1b  = (const float*)d_in[5];
  const float* qkvW  = (const float*)d_in[6];
  const float* qkvb  = (const float*)d_in[7];
  const float* projW = (const float*)d_in[8];
  const float* projb = (const float*)d_in[9];
  const float* ln2g  = (const float*)d_in[10];
  const float* ln2b  = (const float*)d_in[11];
  const float* l1W   = (const float*)d_in[12];
  const float* l1b   = (const float*)d_in[13];
  const float* l2W   = (const float*)d_in[14];
  const float* l2b   = (const float*)d_in[15];
  float* out = (float*)d_out;

  // ---- carve workspace ----
  char* ws = (char*)d_ws;
  size_t off = 0;
  auto carve = [&](size_t bytes) -> char* {
    char* p = ws + off;
    off += (bytes + 255) & ~(size_t)255;
    return p;
  };
  u16*   wq    = (u16*)  carve((size_t)Ee * 3 * Cc * Cc * 2);
  u16*   wp    = (u16*)  carve((size_t)Ee * Cc * Cc * 2);
  u16*   w1    = (u16*)  carve((size_t)Ee * MLPF * Cc * 2);
  u16*   w2    = (u16*)  carve((size_t)Ee * Cc * MLPF * 2);
  float* gate  = (float*)carve((size_t)NTOK * Ee * 4);
  float* stats = (float*)carve(2 * Ee * 4);         // counts[0..7], probsum[8..15]
  u16*   hbf   = (u16*)  carve((size_t)NTOK * Cc * 2);
  float* qkvf  = (float*)carve((size_t)NTOK * 3 * Cc * 4);
  u16*   qbf   = (u16*)  carve((size_t)NTOK * Cc * 2);
  u16*   kbf   = (u16*)  carve((size_t)NTOK * Cc * 2);
  u16*   vtbf  = (u16*)  carve((size_t)NTOK * Cc * 2);
  float* S     = (float*)carve((size_t)Bb * Hh * Nn * Nn * 4);
  float* of    = (float*)carve((size_t)NTOK * Cc * 4);
  u16*   obf   = (u16*)  carve((size_t)NTOK * Cc * 2);
  float* x2    = (float*)carve((size_t)NTOK * Cc * 4);
  u16*   h2bf  = (u16*)  carve((size_t)NTOK * Cc * 2);
  u16*   mbf   = (u16*)  carve((size_t)NTOK * MLPF * 2);
  float* eo    = (float*)carve((size_t)NTOK * Cc * 4);

  const int eb = (NTOK * Cc + 255) / 256;   // 8192 blocks for elementwise [B,N,C]

  // zero combine output + stats
  init_kernel<<<eb, 256, 0, stream>>>(out, stats);

  // weights -> bf16 (then resident in 192MB L2 for all 8 experts)
  f2bf_kernel<<<((long)Ee * 3 * Cc * Cc + 255) / 256, 256, 0, stream>>>(qkvW, wq, (long)Ee * 3 * Cc * Cc);
  f2bf_kernel<<<((long)Ee * Cc * Cc + 255) / 256, 256, 0, stream>>>(projW, wp, (long)Ee * Cc * Cc);
  f2bf_kernel<<<((long)Ee * MLPF * Cc + 255) / 256, 256, 0, stream>>>(l1W, w1, (long)Ee * MLPF * Cc);
  f2bf_kernel<<<((long)Ee * Cc * MLPF + 255) / 256, 256, 0, stream>>>(l2W, w2, (long)Ee * Cc * MLPF);

  // router
  router_kernel<<<(NTOK + 255) / 256, 256, 0, stream>>>(x, rW, rb, noise, gate, stats, stats + Ee);

  for (int e = 0; e < Ee; e++) {
    // LN1 -> h bf16
    ln_kernel<<<NTOK, 256, 0, stream>>>(x, ln1g + (long)e * Cc, ln1b + (long)e * Cc, hbf);
    // qkv = h @ Wqkv^T + b   [4096 x 1536], K=512
    launch_gemm(stream, hbf, Cc, 0, 0, wq + (size_t)e * 3 * Cc * Cc, Cc, 0, 0,
                qkvb + (long)e * 3 * Cc, nullptr, 0, 0, 0,
                qkvf, nullptr, 3 * Cc, 0, 0, NTOK, 3 * Cc, Cc, 1, 1, 1.0f, 0);
    // split into q,k [B,H,N,HD] and v^T [B,H,HD,N], bf16
    split_qkv_kernel<<<eb, 256, 0, stream>>>(qkvf, qbf, kbf, vtbf);
    // S = (q k^T) * sqrt(HD)   batched over B*H: [1024 x 1024], K=64
    launch_gemm(stream, qbf, HDm, (long)Nn * HDm, 0, kbf, HDm, (long)Nn * HDm, 0,
                nullptr, nullptr, 0, 0, 0,
                S, nullptr, Nn, (long)Nn * Nn, 0, Nn, Nn, HDm, Bb * Hh, 1, SCALE_, 0);
    // softmax rows -> P bf16 stored in-place (row stride 2*N u16)
    softmax_kernel<<<Bb * Hh * Nn, 256, 0, stream>>>(S);
    // O = P @ V : batched [1024 x 64], K=1024; output scattered to [B,N,C]
    launch_gemm(stream, (const u16*)S, (long)(2 * Nn), (long)Hh * 2 * Nn * Nn, (long)2 * Nn * Nn,
                vtbf, Nn, (long)Hh * HDm * Nn, (long)HDm * Nn,
                nullptr, nullptr, 0, 0, 0,
                of, nullptr, Cc, (long)Nn * Cc, (long)HDm, Nn, HDm, Nn, Bb * Hh, Hh, 1.0f, 0);
    // proj input to bf16
    f2bf_kernel<<<eb, 256, 0, stream>>>(of, obf, (long)NTOK * Cc);
    // x2 = x + (o @ Wp^T + b)   [4096 x 512], K=512, fused residual
    launch_gemm(stream, obf, Cc, 0, 0, wp + (size_t)e * Cc * Cc, Cc, 0, 0,
                projb + (long)e * Cc, x, Cc, 0, 0,
                x2, nullptr, Cc, 0, 0, NTOK, Cc, Cc, 1, 1, 1.0f, 0);
    // LN2 -> h2 bf16
    ln_kernel<<<NTOK, 256, 0, stream>>>(x2, ln2g + (long)e * Cc, ln2b + (long)e * Cc, h2bf);
    // m = gelu(h2 @ W1^T + b) -> bf16   [4096 x 2048], K=512, fused GELU
    launch_gemm(stream, h2bf, Cc, 0, 0, w1 + (size_t)e * MLPF * Cc, Cc, 0, 0,
                l1b + (long)e * MLPF, nullptr, 0, 0, 0,
                nullptr, mbf, MLPF, 0, 0, NTOK, MLPF, Cc, 1, 1, 1.0f, 1);
    // eo = x2 + (m @ W2^T + b)   [4096 x 512], K=2048, fused residual
    launch_gemm(stream, mbf, MLPF, 0, 0, w2 + (size_t)e * Cc * MLPF, MLPF, 0, 0,
                l2b + (long)e * Cc, x2, Cc, 0, 0,
                eo, nullptr, Cc, 0, 0, NTOK, Cc, MLPF, 1, 1, 1.0f, 0);
    // combine += gate[:,e] * eo
    combine_kernel<<<eb, 256, 0, stream>>>(eo, gate, out, e);
  }

  // balance scalar at out[B*N*C]
  balance_kernel<<<1, 256, 0, stream>>>(rW, stats, out);
}
